// MoBoAligner_51264729645762
// MI455X (gfx1250) — matrix operand order repaired and checked by reading, not compiled, not run on hardware
//
#include <hip/hip_runtime.h>
#include <hip/hip_bf16.h>
#include <math.h>

typedef __attribute__((ext_vector_type(16))) _Float16 v16h;
typedef __attribute__((ext_vector_type(8)))  _Float16 v8h;
typedef __attribute__((ext_vector_type(8)))  float    v8f;

#define LOG_EPS_F (-1000.0f)
#define NEG_INF_F (-1000000000.0f)

// Problem dims (fixed by the reference)
constexpr int Bc = 2, Ic = 64, Jc = 512, DtC = 512, DmC = 80, Ac = 128;

__device__ __forceinline__ float lse2(float a, float b) {
  float m = fmaxf(a, b);
  if (m == -INFINITY) return -INFINITY;
  return m + logf(expf(a - m) + expf(b - m));
}

__device__ __forceinline__ void pack2_to_lds(_Float16* dst, float x, float y) {
  union { _Float16 h[2]; unsigned int u; } pk;
  pk.h[0] = (_Float16)x;
  pk.h[1] = (_Float16)y;
  *(unsigned int*)dst = pk.u;   // 4-byte LDS store
}

// ---------------------------------------------------------------------------
// Generic tiled WMMA GEMM: C[m,n] = sum_k A[m,k] * B(k,n) (+bias[n])
//   A: f32 (M x K), row-major, lda
//   B: if bT==1 -> stored (N x K) row-major, B(k,n)=Bp[n*ldb+k]
//      if bT==0 -> stored (K x N) row-major, B(k,n)=Bp[k*ldb+n]
// f32 -> f16 staged through LDS.  A tile: [m][k], stride 40 halfs (80 B).
// B tile: stored TRANSPOSED [n][k], stride 40 halfs, so WMMA fragments are
// contiguous 16-B chunks -> ds_load_b128, no scalar gathers.
// Workgroup: 256 threads = 8 waves; WG tile 128x128; wave w owns M-rows
// [w*16, w*16+16) over all 128 N columns (8 accumulator tiles).
// ---------------------------------------------------------------------------
#define GEMM_BM 128
#define GEMM_BN 128
#define GEMM_BK 32
#define LDT_P 40    // half-stride (80 B = 16-B multiple, conflict-free)

__global__ void __launch_bounds__(256)
wmma_gemm_kernel(const float* __restrict__ Ap, long sA,
                 const float* __restrict__ Bp, long sB,
                 const float* __restrict__ bias,
                 float* __restrict__ Cp, long sC,
                 int M, int N, int K, int lda, int ldb, int ldc,
                 int bT, int tilesM, int tilesN) {
  __shared__ __align__(16) _Float16 ldsA[GEMM_BM * LDT_P];
  __shared__ __align__(16) _Float16 ldsB[GEMM_BN * LDT_P];  // [n][k]

  const int tpb  = tilesM * tilesN;
  const int bidx = blockIdx.x / tpb;
  const int t    = blockIdx.x - bidx * tpb;
  const int tm   = t / tilesN;
  const int tn   = t - tm * tilesN;
  const int m0   = tm * GEMM_BM;
  const int n0   = tn * GEMM_BN;

  const float* A = Ap + (long)bidx * sA;
  const float* B = Bp + (long)bidx * sB;
  float*       C = Cp + (long)bidx * sC;

  const int tid    = threadIdx.x;
  const int wave   = tid >> 5;
  const int lane   = tid & 31;
  const int lane16 = lane & 15;
  const int hiHalf = (lane >= 16) ? 1 : 0;
  const int mrow   = wave * 16 + lane16;

  v8f acc[8];
#pragma unroll
  for (int nt = 0; nt < 8; ++nt)
#pragma unroll
    for (int r = 0; r < 8; ++r) acc[nt][r] = 0.0f;

  for (int k0 = 0; k0 < K; k0 += GEMM_BK) {
    // --- stage A tile (BM x BK): float2 global reads, packed 2xf16 stores ---
    for (int idx = tid; idx < GEMM_BM * (GEMM_BK / 2); idx += 256) {
      int r  = idx >> 4;          // row in tile
      int c2 = (idx & 15) * 2;    // even k within tile
      int gm = m0 + r, gk = k0 + c2;
      float vx = 0.0f, vy = 0.0f;
      if (gm < M) {
        if (gk + 1 < K) {
          float2 v = *(const float2*)&A[(long)gm * lda + gk];
          vx = v.x; vy = v.y;
        } else if (gk < K) {
          vx = A[(long)gm * lda + gk];
        }
      }
      pack2_to_lds(&ldsA[r * LDT_P + c2], vx, vy);
    }
    // --- stage B tile into [n][k] layout ---
    if (bT) {  // B stored (N x K): k contiguous -> float2 reads, packed stores
      for (int idx = tid; idx < GEMM_BN * (GEMM_BK / 2); idx += 256) {
        int n  = idx >> 4;
        int c2 = (idx & 15) * 2;
        int gn = n0 + n, gk = k0 + c2;
        float vx = 0.0f, vy = 0.0f;
        if (gn < N) {
          if (gk + 1 < K) {
            float2 v = *(const float2*)&B[(long)gn * ldb + gk];
            vx = v.x; vy = v.y;
          } else if (gk < K) {
            vx = B[(long)gn * ldb + gk];
          }
        }
        pack2_to_lds(&ldsB[n * LDT_P + c2], vx, vy);
      }
    } else {   // B stored (K x N): coalesced read over n, scalar transpose store
      for (int idx = tid; idx < GEMM_BK * GEMM_BN; idx += 256) {
        int c = idx >> 7, n = idx & 127;
        int gn = n0 + n, gk = k0 + c;
        float v = (gn < N && gk < K) ? B[(long)gk * ldb + gn] : 0.0f;
        ldsB[n * LDT_P + c] = (_Float16)v;
      }
    }
    // prefetch next A tile stream (emits global_prefetch_b8)
    if (k0 + GEMM_BK < K) {
      int pr = m0 + (tid & 127);
      if (pr < M) __builtin_prefetch(&A[(long)pr * lda + k0 + GEMM_BK], 0, 1);
    }
    __syncthreads();

    // --- A fragment (16x32): lanes 0-15 K={0..7,16..23}, lanes 16-31 +8 ---
    const _Float16* arow = &ldsA[mrow * LDT_P];
    v8h alo = *(const v8h*)(arow + hiHalf * 8);        // ds_load_b128
    v8h ahi = *(const v8h*)(arow + 16 + hiHalf * 8);   // ds_load_b128
    v16h afrag = __builtin_shufflevector(alo, ahi,
        0, 1, 2, 3, 4, 5, 6, 7, 8, 9, 10, 11, 12, 13, 14, 15);

#pragma unroll
    for (int nt = 0; nt < 8; ++nt) {
      // --- B fragment (32x16): lane n=lane%16; K={0..7,16..23}, +8 for hi half
      const _Float16* brow = &ldsB[(nt * 16 + lane16) * LDT_P + hiHalf * 8];
      v8h blo = *(const v8h*)(brow);        // ds_load_b128
      v8h bhi = *(const v8h*)(brow + 16);   // ds_load_b128
      v16h bfrag = __builtin_shufflevector(blo, bhi,
          0, 1, 2, 3, 4, 5, 6, 7, 8, 9, 10, 11, 12, 13, 14, 15);
      acc[nt] = __builtin_amdgcn_wmma_f32_16x16x32_f16(
          false, afrag, false, bfrag, (short)0, acc[nt], false, false);
    }
    __syncthreads();
  }

  // --- store: lane -> N=lane%16, VGPR r -> M=r (+8 for hi half-wave)
#pragma unroll
  for (int nt = 0; nt < 8; ++nt) {
    int n = n0 + nt * 16 + lane16;
#pragma unroll
    for (int r = 0; r < 8; ++r) {
      int m = m0 + wave * 16 + r + hiHalf * 8;
      if (m < M && n < N) {
        float v = acc[nt][r];
        if (bias) v += bias[n];
        C[(long)m * ldc + n] = v;
      }
    }
  }
}

// ---------------------------------------------------------------------------
// Energy + suffix-logsumexp T.  One block per (b,i), 512 threads (j).
//   E[b,i,j] = sigmoid( sum_a tanh(pm[b,j,a]+pt[b,i,a])*v_w[a] + v_b + 2*noise )
//   masked rows -> LOG_EPS.  T[b,i,j] = log( sum_{j'>=j} exp(E) ),
//   T[b,i,J] = LOG_EPS + log(J)  (matches reference all-LOG_EPS column).
// ---------------------------------------------------------------------------
__global__ void __launch_bounds__(512)
energy_kernel(const float* __restrict__ pm, const float* __restrict__ pt,
              const float* __restrict__ noise,
              const unsigned char* __restrict__ tmask,
              const unsigned char* __restrict__ mmask,
              const float* __restrict__ v_w, const float* __restrict__ v_b,
              float* __restrict__ E, float* __restrict__ T) {
  const int bi = blockIdx.x;           // b*I + i
  const int b  = bi / Ic;
  const int i  = bi - b * Ic;
  const int j  = threadIdx.x;          // 0..J-1

  __shared__ float spt[Ac];
  __shared__ float svw[Ac];
  __shared__ float ssum[Jc];

  if (j < Ac) {
    spt[j] = pt[(long)bi * Ac + j];
    svw[j] = v_w[j];
  }
  __syncthreads();

  const float4* pmrow4 = (const float4*)(pm + ((long)b * Jc + j) * Ac);
  float acc = 0.0f;
#pragma unroll 4
  for (int a4 = 0; a4 < Ac / 4; ++a4) {
    float4 p = pmrow4[a4];             // global_load_b128
    int a = a4 * 4;
    acc += tanhf(p.x + spt[a + 0]) * svw[a + 0];
    acc += tanhf(p.y + spt[a + 1]) * svw[a + 1];
    acc += tanhf(p.z + spt[a + 2]) * svw[a + 2];
    acc += tanhf(p.w + spt[a + 3]) * svw[a + 3];
  }

  float x = acc + v_b[0] + noise[(long)bi * Jc + j] * 2.0f;
  float e = 1.0f / (1.0f + expf(-x));
  bool valid = (tmask[b * Ic + i] != 0) && (mmask[b * Jc + j] != 0);
  float em = valid ? e : LOG_EPS_F;
  E[(long)bi * Jc + j] = em;

  // suffix sum of exp(em): E bounded in (0,1), no max-shift needed
  ssum[j] = expf(em);
  __syncthreads();
  for (int off = 1; off < Jc; off <<= 1) {
    float v = ssum[j];
    if (j + off < Jc) v += ssum[j + off];
    __syncthreads();
    ssum[j] = v;
    __syncthreads();
  }
  float* Trow = T + (long)bi * (Jc + 1);
  Trow[j] = logf(ssum[j]);
  if (j == 0) Trow[Jc] = LOG_EPS_F + logf((float)Jc);
}

// ---------------------------------------------------------------------------
// DP over text rows.  One block per batch, 512 threads (column jcol).
// Produces prob[b,i,j] = Bij[b,i,j], rows 0..I-1, cols 0..J-1.
// Per row: two Hillis-Steele lse scans replace the O(J^2) reference lse.
// ---------------------------------------------------------------------------
__global__ void __launch_bounds__(512)
dp_kernel(const float* __restrict__ E, const float* __restrict__ T,
          const unsigned char* __restrict__ tmask,
          const unsigned char* __restrict__ mmask,
          float* __restrict__ prob) {
  const int b = blockIdx.x;
  const int j = threadIdx.x;

  __shared__ float prev[Jc];
  __shared__ float pre[Jc];
  __shared__ float suf[Jc];
  __shared__ float contrib[Jc];
  __shared__ int sJm;

  if (j == 0) {  // Jm = max_b( sum(mel_mask) - sum(text_mask) )
    int jm = -2147483647;
    for (int bb = 0; bb < Bc; ++bb) {
      int ms = 0, ts = 0;
      for (int k = 0; k < Jc; ++k) ms += mmask[bb * Jc + k] ? 1 : 0;
      for (int k = 0; k < Ic; ++k) ts += tmask[bb * Ic + k] ? 1 : 0;
      int d = ms - ts;
      if (d > jm) jm = d;
    }
    sJm = jm;
  }
  float p0 = (j == 0) ? 0.0f : NEG_INF_F;
  prev[j] = p0;
  prob[((long)b * Ic + 0) * Jc + j] = p0;
  __syncthreads();
  const int Jm = sJm;

  for (int i = 1; i < Ic; ++i) {
    const float* Trow = T + ((long)b * Ic + (i - 1)) * (Jc + 1);
    const float* Erow = E + ((long)b * Ic + (i - 1)) * Jc;

    float q = prev[j] - Trow[j + 1];
    pre[j] = q;
    suf[j] = q;
    __syncthreads();
    for (int off = 1; off < Jc; off <<= 1) {
      float vp = pre[j];
      if (j >= off) vp = lse2(pre[j - off], vp);
      float vs = suf[j];
      if (j + off < Jc) vs = lse2(vs, suf[j + off]);
      __syncthreads();
      pre[j] = vp;
      suf[j] = vs;
      __syncthreads();
    }
    float P = (j == 0) ? -INFINITY : pre[j - 1];         // exclusive prefix
    contrib[j] = lse2(Erow[j] + P, LOG_EPS_F + suf[j]);  // big + tiny paths
    __syncthreads();

    float v;
    if (j == 0) {
      v = NEG_INF_F;                       // row[:,0] = NEG_INF
    } else {
      v = contrib[j - 1];                  // row[jcol] = contrib[jcol-1]
      if (!(j >= i && j < Jm + i + 2)) v = NEG_INF_F;  // window mask
    }
    prev[j] = v;
    prob[((long)b * Ic + i) * Jc + j] = v;
    __syncthreads();
  }
}

// ---------------------------------------------------------------------------
// soft[b,i,k] = lse_j( prob[j] + (j<k ? T[k]-T[j+1] : LOG_EPS) )
//            = lse( T[k] + prefixLSE_{j<k}(prob[j]-T[j+1]),
//                   LOG_EPS + suffixLSE_{j>=k}(prob[j]) )
// One block per (b,i), 512 threads.
// ---------------------------------------------------------------------------
__global__ void __launch_bounds__(512)
soft_kernel(const float* __restrict__ prob, const float* __restrict__ T,
            float* __restrict__ soft) {
  const int bi = blockIdx.x;
  const int j  = threadIdx.x;

  __shared__ float pre[Jc];
  __shared__ float suf[Jc];

  const float* Prow = prob + (long)bi * Jc;
  const float* Trow = T + (long)bi * (Jc + 1);
  float p = Prow[j];
  pre[j] = p - Trow[j + 1];
  suf[j] = p;
  __syncthreads();
  for (int off = 1; off < Jc; off <<= 1) {
    float vp = pre[j];
    if (j >= off) vp = lse2(pre[j - off], vp);
    float vs = suf[j];
    if (j + off < Jc) vs = lse2(vs, suf[j + off]);
    __syncthreads();
    pre[j] = vp;
    suf[j] = vs;
    __syncthreads();
  }
  float P = (j == 0) ? -INFINITY : pre[j - 1];
  soft[(long)bi * Jc + j] = lse2(Trow[j] + P, LOG_EPS_F + suf[j]);
}

// ---------------------------------------------------------------------------
// Wt[b, j, i] = exp(soft[b,i,j]) * align_mask  (A-matrix for expanded GEMM)
// ---------------------------------------------------------------------------
__global__ void weight_kernel(const float* __restrict__ soft,
                              const unsigned char* __restrict__ tmask,
                              const unsigned char* __restrict__ mmask,
                              float* __restrict__ Wt) {
  int idx = blockIdx.x * blockDim.x + threadIdx.x;  // over B*I*J
  if (idx >= Bc * Ic * Jc) return;
  int j  = idx % Jc;
  int bi = idx / Jc;
  int b  = bi / Ic;
  int i  = bi - b * Ic;
  float w = expf(soft[idx]);
  if (!(tmask[b * Ic + i] && mmask[b * Jc + j])) w = 0.0f;
  Wt[((long)b * Jc + j) * Ic + i] = w;
}

// ---------------------------------------------------------------------------
extern "C" void kernel_launch(void* const* d_in, const int* in_sizes, int n_in,
                              void* d_out, int out_size, void* d_ws, size_t ws_size,
                              hipStream_t stream) {
  (void)in_sizes; (void)n_in; (void)out_size; (void)ws_size;

  const float* text   = (const float*)d_in[0];  // (B,I,Dt)
  const float* mel    = (const float*)d_in[1];  // (B,J,Dm)
  const float* noise  = (const float*)d_in[2];  // (B,I,J)
  const unsigned char* tmask = (const unsigned char*)d_in[3];  // (B,I) bool
  const unsigned char* mmask = (const unsigned char*)d_in[4];  // (B,J) bool
  const float* mel_w  = (const float*)d_in[5];  // (A,Dm)
  const float* mel_b  = (const float*)d_in[6];  // (A,)
  const float* text_w = (const float*)d_in[7];  // (A,Dt)
  const float* v_w    = (const float*)d_in[8];  // (A,)
  const float* v_b    = (const float*)d_in[9];  // (1,)

  float* out      = (float*)d_out;
  float* soft     = out;                         // (B,I,J)
  float* expanded = out + (long)Bc * Ic * Jc;    // (B,J,Dt)

  float* ws   = (float*)d_ws;
  float* pm   = ws; ws += (long)Bc * Jc * Ac;        // (B,J,A)
  float* pt   = ws; ws += (long)Bc * Ic * Ac;        // (B,I,A)
  float* E    = ws; ws += (long)Bc * Ic * Jc;        // (B,I,J)
  float* T    = ws; ws += (long)Bc * Ic * (Jc + 1);  // (B,I,J+1)
  float* prob = ws; ws += (long)Bc * Ic * Jc;        // (B,I,J)
  float* Wt   = ws; ws += (long)Bc * Jc * Ic;        // (B,J,I)

  // pm = mel @ mel_w^T + mel_b   (M=J, N=A, K=Dm, B stored (N,K))
  {
    int tM = (Jc + GEMM_BM - 1) / GEMM_BM, tN = (Ac + GEMM_BN - 1) / GEMM_BN;
    wmma_gemm_kernel<<<Bc * tM * tN, 256, 0, stream>>>(
        mel, (long)Jc * DmC, mel_w, 0L, mel_b, pm, (long)Jc * Ac,
        Jc, Ac, DmC, DmC, DmC, Ac, /*bT=*/1, tM, tN);
  }
  // pt = text @ text_w^T         (M=I, N=A, K=Dt, B stored (N,K))
  {
    int tM = (Ic + GEMM_BM - 1) / GEMM_BM, tN = (Ac + GEMM_BN - 1) / GEMM_BN;
    wmma_gemm_kernel<<<Bc * tM * tN, 256, 0, stream>>>(
        text, (long)Ic * DtC, text_w, 0L, nullptr, pt, (long)Ic * Ac,
        Ic, Ac, DtC, DtC, DtC, Ac, /*bT=*/1, tM, tN);
  }

  energy_kernel<<<Bc * Ic, Jc, 0, stream>>>(pm, pt, noise, tmask, mmask, v_w, v_b, E, T);
  dp_kernel<<<Bc, Jc, 0, stream>>>(E, T, tmask, mmask, prob);
  soft_kernel<<<Bc * Ic, Jc, 0, stream>>>(prob, T, soft);
  weight_kernel<<<(Bc * Ic * Jc + 255) / 256, 256, 0, stream>>>(soft, tmask, mmask, Wt);

  // expanded[b] = Wt[b](JxI) @ text[b](IxDt)   (M=J, N=Dt, K=I, B stored (K,N))
  {
    int tM = (Jc + GEMM_BM - 1) / GEMM_BM, tN = (DtC + GEMM_BN - 1) / GEMM_BN;
    wmma_gemm_kernel<<<Bc * tM * tN, 256, 0, stream>>>(
        Wt, (long)Jc * Ic, text, (long)Ic * DtC, nullptr, expanded, (long)Jc * DtC,
        Jc, DtC, Ic, Ic, DtC, DtC, /*bT=*/0, tM, tN);
  }
}
